// WeightedGCNModel_v1_78357383349013
// MI455X (gfx1250) — matrix operand order repaired and checked by reading, not compiled, not run on hardware
//
#include <hip/hip_runtime.h>
#include <hip/hip_bf16.h>

// ---------------------------------------------------------------------------
// WeightedGCN on MI455X (gfx1250, wave32).
//  - Dense x@W through v_wmma_f32_16x16x32_bf16 (bf16 in, f32 accumulate).
//  - Sparse aggregation restructured as CSR (group edges by dst once, reuse
//    for both layers) -> gather-only accumulation in registers, no float
//    atomics in the hot path; fuses self-loop + bias + ReLU.
//    51MB feature matrix is L2-resident (192MB L2), so random row gathers
//    resolve in L2.
// ---------------------------------------------------------------------------

#define GCN_HID 128

typedef __bf16 v16bf __attribute__((ext_vector_type(16)));
typedef __bf16 v8bf  __attribute__((ext_vector_type(8)));
typedef float  v8f   __attribute__((ext_vector_type(8)));

__device__ __forceinline__ unsigned short f2bf_bits(float f) {
    unsigned u = __builtin_bit_cast(unsigned, f);
    u = (u + 0x7FFFu + ((u >> 16) & 1u)) >> 16;      // round-to-nearest-even
    return (unsigned short)u;
}
__device__ __forceinline__ __bf16 f2bf(float f) {
    unsigned short s = f2bf_bits(f);
    return __builtin_bit_cast(__bf16, s);
}
__device__ __forceinline__ void atomAddF(float* p, float v) {
    __hip_atomic_fetch_add(p, v, __ATOMIC_RELAXED, __HIP_MEMORY_SCOPE_AGENT);
}

// ---- fills -----------------------------------------------------------------
__global__ void k_fill(float* p, float v, int n) {
    int t = blockIdx.x * blockDim.x + threadIdx.x;
    if (t < n) p[t] = v;
}
__global__ void k_filli(int* p, int v, int n) {
    int t = blockIdx.x * blockDim.x + threadIdx.x;
    if (t < n) p[t] = v;
}

// ---- degree / normalization -----------------------------------------------
__global__ void k_deg_edges(const int* __restrict__ dst,
                            const float* __restrict__ ew,
                            float* __restrict__ deg, int nE) {
    int t = blockIdx.x * blockDim.x + threadIdx.x;
    if (t < nE) atomAddF(&deg[dst[t]], ew[t]);
}
__global__ void k_dinv(const float* __restrict__ deg, float* __restrict__ dinv, int n) {
    int t = blockIdx.x * blockDim.x + threadIdx.x;
    if (t < n) { float d = deg[t]; dinv[t] = d > 0.f ? rsqrtf(d) : 0.f; }
}

// ---- CSR construction: histogram -> 2-level exclusive scan -> reorder -----
__global__ void k_hist(const int* __restrict__ dst, int* __restrict__ ecnt, int nE) {
    int t = blockIdx.x * blockDim.x + threadIdx.x;
    if (t < nE) atomicAdd(&ecnt[dst[t]], 1);
}
// block of 256 threads scans 1024 counts -> exclusive prefix + block total
__global__ __launch_bounds__(256) void k_scan_local(
    const int* __restrict__ cnt, int* __restrict__ ex, int* __restrict__ bsum, int n) {
    __shared__ int sh[256];
    int t = threadIdx.x;
    int base = blockIdx.x * 1024 + t * 4;
    int v0 = base + 0 < n ? cnt[base + 0] : 0;
    int v1 = base + 1 < n ? cnt[base + 1] : 0;
    int v2 = base + 2 < n ? cnt[base + 2] : 0;
    int v3 = base + 3 < n ? cnt[base + 3] : 0;
    int s = v0 + v1 + v2 + v3;
    sh[t] = s;
    __syncthreads();
    for (int off = 1; off < 256; off <<= 1) {        // Hillis-Steele inclusive
        int x = (t >= off) ? sh[t - off] : 0;
        __syncthreads();
        sh[t] += x;
        __syncthreads();
    }
    int exc = sh[t] - s;                              // exclusive prefix
    if (base + 0 < n) ex[base + 0] = exc;
    if (base + 1 < n) ex[base + 1] = exc + v0;
    if (base + 2 < n) ex[base + 2] = exc + v0 + v1;
    if (base + 3 < n) ex[base + 3] = exc + v0 + v1 + v2;
    if (t == 255) bsum[blockIdx.x] = sh[255];
}
__global__ __launch_bounds__(128) void k_scan_bsum(int* __restrict__ bsum, int nb) {
    __shared__ int sh[128];
    int t = threadIdx.x;
    sh[t] = (t < nb) ? bsum[t] : 0;
    __syncthreads();
    int acc = 0;
    for (int j = 0; j < t; ++j) acc += sh[j];         // nb <= 128, tiny
    if (t < nb) bsum[t] = acc;
}
__global__ void k_scan_add(int* __restrict__ ex, const int* __restrict__ bsum,
                           int n, int nE) {
    int t = blockIdx.x * blockDim.x + threadIdx.x;
    if (t < n) ex[t] += bsum[t >> 10];
    if (t == 0) ex[n] = nE;                           // rowptr[N] = E
}
__global__ void k_copyi(const int* __restrict__ a, int* __restrict__ b, int n) {
    int t = blockIdx.x * blockDim.x + threadIdx.x;
    if (t < n) b[t] = a[t];
}
// place each edge into its dst bucket; precompute the GCN norm coefficient
__global__ void k_reorder(const int* __restrict__ src, const int* __restrict__ dst,
                          const float* __restrict__ ew, const float* __restrict__ dinv,
                          int* __restrict__ cursor, int* __restrict__ esrc,
                          float* __restrict__ coef, int nE) {
    int t = blockIdx.x * blockDim.x + threadIdx.x;
    if (t >= nE) return;
    int s = src[t], d = dst[t];
    int pos = atomicAdd(&cursor[d], 1);
    esrc[pos] = s;
    coef[pos] = dinv[s] * ew[t] * dinv[d];
}

// ---- pack 128x128 fp32 weight into WMMA B-fragment order (bf16) -----------
__global__ void k_prep_wfrag(const float* __restrict__ W, unsigned short* __restrict__ wfrag) {
    int t = blockIdx.x * blockDim.x + threadIdx.x;     // 8*4*32*16 = 16384
    int e    = t & 15;
    int lane = (t >> 4) & 31;
    int ks   = (t >> 9) & 3;
    int ct   = (t >> 11) & 7;
    int K    = ks * 32 + 16 * (lane >> 4) + e;
    int col  = ct * 16 + (lane & 15);
    wfrag[t] = f2bf_bits(W[K * GCN_HID + col]);
}

// ---- tiled GEMM: out[rows,128] = gather(rows_src, idx) @ W  (bf16 WMMA) ----
__global__ __launch_bounds__(256) void k_gemm_bf16(
    const float* __restrict__ rows_src,
    const int*   __restrict__ idx,          // row gather indices, or nullptr
    const unsigned short* __restrict__ wfrag,
    float* __restrict__ out, int nrows) {
    __shared__ __align__(16) __bf16 As[16 * GCN_HID];

    const int tid  = threadIdx.x;
    const int tile = blockIdx.x;
    {
        int r = tid >> 4;
        int c = (tid & 15) * 8;
        int grow = tile * 16 + r;
        const float* p = rows_src + (long)(idx ? idx[grow] : grow) * GCN_HID + c;
        float4 v0 = *(const float4*)(p);
        float4 v1 = *(const float4*)(p + 4);
        __bf16* d = &As[r * GCN_HID + c];
        d[0] = f2bf(v0.x); d[1] = f2bf(v0.y); d[2] = f2bf(v0.z); d[3] = f2bf(v0.w);
        d[4] = f2bf(v1.x); d[5] = f2bf(v1.y); d[6] = f2bf(v1.z); d[7] = f2bf(v1.w);
    }
    __syncthreads();

    const int wave = tid >> 5;
    const int lane = tid & 31;
    const int m    = lane & 15;
    const int half = lane >> 4;

    v8f acc = {};
#pragma unroll
    for (int ks = 0; ks < 4; ++ks) {
        const int kk = ks * 32;
        v8bf alo = *(const v8bf*)&As[m * GCN_HID + kk + 16 * half];
        v8bf ahi = *(const v8bf*)&As[m * GCN_HID + kk + 16 * half + 8];
        v16bf a;
#pragma unroll
        for (int i = 0; i < 8; ++i) { a[i] = alo[i]; a[8 + i] = ahi[i]; }
        v16bf b = *(const v16bf*)(wfrag + (((wave * 4 + ks) * 32 + lane) << 4));
        acc = __builtin_amdgcn_wmma_f32_16x16x32_bf16(
                  false, a, false, b, (short)0, acc, false, false);
    }
    float* o = out + (long)(tile * 16 + 8 * half) * GCN_HID + wave * 16 + m;
#pragma unroll
    for (int r = 0; r < 8; ++r) o[(long)r * GCN_HID] = acc[r];
}

// ---- gather-only aggregation: one wave32 per dst node ----------------------
// h[d] = relu( bias + dinv[d]^2 * xw[d] + sum_{e in CSR[d]} coef[e]*xw[src[e]] )
__global__ __launch_bounds__(256) void k_aggregate(
    const float* __restrict__ xw, const int* __restrict__ rowptr,
    const int* __restrict__ esrc, const float* __restrict__ coef,
    const float* __restrict__ dinv, const float* __restrict__ bias,
    float* __restrict__ h, int n) {
    int node = blockIdx.x * (blockDim.x >> 5) + (threadIdx.x >> 5);
    if (node >= n) return;
    int lane = threadIdx.x & 31;
    int q = lane * 4;

    float di = dinv[node];
    float c0 = di * di;                               // self-loop norm (w=1)
    float4 v = *(const float4*)(xw + (long)node * GCN_HID + q);
    float a0 = v.x * c0, a1 = v.y * c0, a2 = v.z * c0, a3 = v.w * c0;

    int kbeg = rowptr[node], kend = rowptr[node + 1];
    for (int k0 = kbeg; k0 < kend; k0 += 32) {
        int myk = k0 + lane;
        int   sl = (myk < kend) ? esrc[myk] : 0;      // cooperative meta load
        float cl = (myk < kend) ? coef[myk] : 0.f;
        int cnt = kend - k0; if (cnt > 32) cnt = 32;
        for (int j = 0; j < cnt; ++j) {
            int   s = __shfl(sl, j, 32);
            float c = __shfl(cl, j, 32);
            float4 u = *(const float4*)(xw + (long)s * GCN_HID + q);
            a0 = fmaf(u.x, c, a0); a1 = fmaf(u.y, c, a1);
            a2 = fmaf(u.z, c, a2); a3 = fmaf(u.w, c, a3);
        }
    }
    float4 r;
    r.x = a0 + bias[q + 0]; r.y = a1 + bias[q + 1];
    r.z = a2 + bias[q + 2]; r.w = a3 + bias[q + 3];
    r.x = r.x > 0.f ? r.x : 0.f; r.y = r.y > 0.f ? r.y : 0.f;
    r.z = r.z > 0.f ? r.z : 0.f; r.w = r.w > 0.f ? r.w : 0.f;
    *(float4*)(h + (long)node * GCN_HID + q) = r;
}

// ---- global mean pool ------------------------------------------------------
__global__ void k_pool(const float* __restrict__ h, const int* __restrict__ batch,
                       float* __restrict__ pooled, float* __restrict__ cnt, int n) {
    int t = blockIdx.x * blockDim.x + threadIdx.x;     // n*32 threads
    if (t >= n * 32) return;
    int node = t >> 5, q = (t & 31) * 4;
    int b = batch[node];
    float4 v = *(const float4*)(h + (long)node * GCN_HID + q);
    float* o = pooled + (long)b * GCN_HID + q;
    atomAddF(o + 0, v.x); atomAddF(o + 1, v.y);
    atomAddF(o + 2, v.z); atomAddF(o + 3, v.w);
    if ((t & 31) == 0) atomAddF(&cnt[b], 1.f);
}

// ---- MLP head + softmax ----------------------------------------------------
__global__ __launch_bounds__(128) void k_head(
    const float* __restrict__ pooled, const float* __restrict__ cnt,
    const float* __restrict__ W1, const float* __restrict__ b1,
    const float* __restrict__ W2, const float* __restrict__ b2,
    float* __restrict__ out, int nB) {
    int g = threadIdx.x;
    if (g >= nB) return;
    float c = cnt[g]; c = c > 1.f ? c : 1.f;
    float inv = 1.f / c;
    float hbuf[GCN_HID / 2];
#pragma unroll 4
    for (int k = 0; k < GCN_HID / 2; ++k) {
        float a = b1[k];
        for (int j = 0; j < GCN_HID; ++j)
            a += pooled[g * GCN_HID + j] * inv * W1[j * (GCN_HID / 2) + k];
        hbuf[k] = a > 0.f ? a : 0.f;
    }
    float l0 = b2[0], l1 = b2[1];
    for (int k = 0; k < GCN_HID / 2; ++k) { l0 += hbuf[k] * W2[2 * k]; l1 += hbuf[k] * W2[2 * k + 1]; }
    float mx = l0 > l1 ? l0 : l1;
    float e0 = __expf(l0 - mx), e1 = __expf(l1 - mx);
    float s = e0 + e1;
    out[g * 2 + 0] = e0 / s;
    out[g * 2 + 1] = e1 / s;
}

extern "C" void kernel_launch(void* const* d_in, const int* in_sizes, int n_in,
                              void* d_out, int out_size, void* d_ws, size_t ws_size,
                              hipStream_t stream) {
    const int N = in_sizes[0];                 // 100000
    const int E = in_sizes[2];                 // 1600000
    const int B = out_size / 2;                // 128

    const int*   x_idx = (const int*)d_in[0];
    const int*   src   = (const int*)d_in[1];
    const int*   dst   = src + E;
    const float* ew    = (const float*)d_in[2];
    const int*   batch = (const int*)d_in[3];
    const float* emb   = (const float*)d_in[4];
    const float* W1    = (const float*)d_in[5];
    const float* b1    = (const float*)d_in[6];
    const float* W2    = (const float*)d_in[7];
    const float* b2    = (const float*)d_in[8];
    const float* mW1   = (const float*)d_in[9];
    const float* mb1   = (const float*)d_in[10];
    const float* mW2   = (const float*)d_in[11];
    const float* mb2   = (const float*)d_in[12];
    float* out = (float*)d_out;

    // ---- workspace layout
    float* bufA   = (float*)d_ws;                       // xw   [N,128]
    float* bufB   = bufA + (size_t)N * GCN_HID;         // h    [N,128]
    float* deg    = bufB + (size_t)N * GCN_HID;         // [N]
    float* dinv   = deg + N;                            // [N]
    float* pooled = dinv + N;                           // [B,128]
    float* cnt    = pooled + (size_t)B * GCN_HID;       // [B]
    unsigned short* wf1 = (unsigned short*)(cnt + B);   // 16384 bf16 (32B aligned)
    unsigned short* wf2 = wf1 + 8 * 4 * 32 * 16;
    int*   rowptr = (int*)(wf2 + 8 * 4 * 32 * 16);      // [N+1]
    int*   cursor = rowptr + (N + 1);                   // [N]
    int*   ecnt   = cursor + N;                         // [N]
    int*   bsum   = ecnt + N;                           // [128]
    int*   esrc   = bsum + 128;                         // [E]
    float* coef   = (float*)(esrc + E);                 // [E]

    const int TB = 256;
    const int nodeBlks = (N + TB - 1) / TB;
    const int edgeBlks = (E + TB - 1) / TB;
    const int node32   = (N * 32 + TB - 1) / TB;
    const int waveNode = (N + 7) / 8;                   // one wave per node
    const int rowTiles = N / 16;                        // N % 16 == 0
    const int nbScan   = (N + 1023) / 1024;             // <= 128

    // ---- degree / dinv
    k_fill<<<nodeBlks, TB, 0, stream>>>(deg, 1.0f, N);  // self-loop weight 1
    k_deg_edges<<<edgeBlks, TB, 0, stream>>>(dst, ew, deg, E);
    k_dinv<<<nodeBlks, TB, 0, stream>>>(deg, dinv, N);

    // ---- CSR build (reused by both layers)
    k_filli<<<nodeBlks, TB, 0, stream>>>(ecnt, 0, N);
    k_hist<<<edgeBlks, TB, 0, stream>>>(dst, ecnt, E);
    k_scan_local<<<nbScan, TB, 0, stream>>>(ecnt, rowptr, bsum, N);
    k_scan_bsum<<<1, 128, 0, stream>>>(bsum, nbScan);
    k_scan_add<<<nodeBlks, TB, 0, stream>>>(rowptr, bsum, N, E);
    k_copyi<<<nodeBlks, TB, 0, stream>>>(rowptr, cursor, N);
    k_reorder<<<edgeBlks, TB, 0, stream>>>(src, dst, ew, dinv, cursor, esrc, coef, E);

    // ---- weight fragments (bf16, WMMA B order)
    k_prep_wfrag<<<64, TB, 0, stream>>>(W1, wf1);
    k_prep_wfrag<<<64, TB, 0, stream>>>(W2, wf2);

    // ---- layer 1: embed-gather + WMMA GEMM, fused aggregate+bias+relu
    k_gemm_bf16<<<rowTiles, TB, 0, stream>>>(emb, x_idx, wf1, bufA, N);
    k_aggregate<<<waveNode, TB, 0, stream>>>(bufA, rowptr, esrc, coef, dinv, b1, bufB, N);

    // ---- layer 2
    k_gemm_bf16<<<rowTiles, TB, 0, stream>>>(bufB, nullptr, wf2, bufA, N);
    k_aggregate<<<waveNode, TB, 0, stream>>>(bufA, rowptr, esrc, coef, dinv, b2, bufB, N);

    // ---- mean pool + head
    k_fill<<<(B * GCN_HID + TB - 1) / TB, TB, 0, stream>>>(pooled, 0.f, B * GCN_HID);
    k_fill<<<1, TB, 0, stream>>>(cnt, 0.f, B);
    k_pool<<<node32, TB, 0, stream>>>(bufB, batch, pooled, cnt, N);
    k_head<<<1, 128, 0, stream>>>(pooled, cnt, mW1, mb1, mW2, mb2, out, B);
}
